// MultiViewUniMatch_2619930051043
// MI455X (gfx1250) — matrix operand order repaired and checked by reading, not compiled, not run on hardware
//
#include <hip/hip_runtime.h>

typedef __attribute__((ext_vector_type(16))) __bf16 v16bf;
typedef __attribute__((ext_vector_type(8)))  __bf16 v8bf;
typedef __attribute__((ext_vector_type(8)))  float  v8f;

#define BB   8
#define MMV  3
#define HH   96
#define CCH  128
#define WS   24
#define LL   576
#define SHF  12
#define NWIN 128
#define QT   64
#define NQT  9
#define KT   32

#define KSTRIDE 136   // bf16 per K row (128 + 8 pad) -> 272B, 16B aligned, bank-spread
#define VSTRIDE 40    // bf16 per V^T row (32 + 8)    -> 80B
#define PSTRIDE 36    // f32 per P row (32 + 4)       -> 144B

__device__ __forceinline__ int region_of(int wy, int wx, int idx) {
    int iy = idx / WS, ix = idx % WS;
    int y = wy * WS + iy;
    int x = wx * WS + ix;
    int yr = (y < HH - WS) ? 0 : ((y < HH - SHF) ? 1 : 2);
    int xr = (x < HH - WS) ? 0 : ((x < HH - SHF) ? 1 : 2);
    return yr * 3 + xr;
}

__device__ __forceinline__ v16bf ld_bfrag16(const __bf16* p) {
    // 16 consecutive bf16 from LDS (two 16B reads)
    v8bf lo = *(const v8bf*)p;
    v8bf hi = *(const v8bf*)(p + 8);
    v16bf r;
#pragma unroll
    for (int i = 0; i < 8; ++i) { r[i] = lo[i]; r[i + 8] = hi[i]; }
    return r;
}

__global__ __launch_bounds__(128) void mvum_attn_kernel(
    const float* __restrict__ qin, const float* __restrict__ kin,
    const float* __restrict__ vin, float* __restrict__ out)
{
    __shared__ __bf16 Kbuf[KT * KSTRIDE];      // [kv][c] bf16
    __shared__ __bf16 VTbuf[CCH * VSTRIDE];    // [c][kv] bf16 (transposed V)
    __shared__ float  Pbuf[4 * 16 * PSTRIDE];  // per-wave P bounce [q][kv] f32

    const int blk  = blockIdx.x;
    const int win  = blk / NQT;
    const int qt   = blk % NQT;
    const int b    = win >> 4;
    const int widx = win & 15;
    const int wy   = widx >> 2;
    const int wx   = widx & 3;

    const int tid  = threadIdx.x;
    const int wave = tid >> 5;
    const int lane = tid & 31;
    const int half = lane >> 4;
    const int ln16 = lane & 15;

    const float SCALE  = 0.08838834764831845f * 1.4426950408889634f; // (1/sqrt(C)) * log2(e)
    const float MASKB2 = -100.0f * 1.4426950408889634f;              // mask in base-2 domain

    // ---------------- Q fragments (A layout), scaled, once ----------------
    const int qb = qt * QT + wave * 16;       // window-local q base for this wave
    {
    }
    int aqrow = qb + ln16;                    // this lane's A-matrix row
    int qiy = aqrow / WS, qix = aqrow % WS;
    int qsy = (wy * WS + qiy + SHF) % HH;
    int qsx = (wx * WS + qix + SHF) % HH;
    const float* qg = qin + ((size_t)b * (HH * HH) + (size_t)qsy * HH + qsx) * CCH;

    v16bf qfrag[4];
#pragma unroll
    for (int ck = 0; ck < 4; ++ck) {
        const float* p0 = qg + ck * 32 + 16 * half;       // K(e)=e+16h, e=0..7
        const float* p1 = qg + ck * 32 + 8 + 16 * half;   // K(e)=e+16h, e=8..15
        float4 a0 = *(const float4*)(p0);
        float4 a1 = *(const float4*)(p0 + 4);
        float4 b0 = *(const float4*)(p1);
        float4 b1 = *(const float4*)(p1 + 4);
        float t[16] = { a0.x, a0.y, a0.z, a0.w, a1.x, a1.y, a1.z, a1.w,
                        b0.x, b0.y, b0.z, b0.w, b1.x, b1.y, b1.z, b1.w };
        v16bf f;
#pragma unroll
        for (int e = 0; e < 16; ++e) f[e] = (__bf16)(t[e] * SCALE);
        qfrag[ck] = f;
    }

    // per-lane q-row regions for rows r + 8*half of this wave's tile
    int regq[8];
#pragma unroll
    for (int r = 0; r < 8; ++r)
        regq[r] = region_of(wy, wx, qb + r + 8 * half);

    // ---------------- accumulators ----------------
    v8f acc[8];
#pragma unroll
    for (int t = 0; t < 8; ++t) {
#pragma unroll
        for (int e = 0; e < 8; ++e) acc[t][e] = 0.0f;
    }
    float mrun[8], lrun[8];
#pragma unroll
    for (int r = 0; r < 8; ++r) { mrun[r] = -3.0e38f; lrun[r] = 0.0f; }

    // ---------------- main loop over views and kv chunks ----------------
    for (int m = 0; m < MMV; ++m) {
        const float* kbase = kin + (size_t)(b * MMV + m) * (HH * HH) * CCH;
        const float* vbase = vin + (size_t)(b * MMV + m) * (HH * HH) * CCH;

        for (int l0 = 0; l0 < LL; l0 += KT) {
            __syncthreads();
            // ---- stage K chunk [32 x 128] -> bf16 [kv][c] ----
            {
                int row = tid >> 2;
                int c0  = (tid & 3) * 32;
                int l   = l0 + row;
                int iy = l / WS, ix = l % WS;
                int sy = (wy * WS + iy + SHF) % HH;
                int sx = (wx * WS + ix + SHF) % HH;
                const float* g = kbase + ((size_t)sy * HH + sx) * CCH + c0;
#pragma unroll
                for (int j = 0; j < 4; ++j) {
                    float4 a = *(const float4*)(g + j * 8);
                    float4 c = *(const float4*)(g + j * 8 + 4);
                    v8bf pk;
                    pk[0] = (__bf16)a.x; pk[1] = (__bf16)a.y;
                    pk[2] = (__bf16)a.z; pk[3] = (__bf16)a.w;
                    pk[4] = (__bf16)c.x; pk[5] = (__bf16)c.y;
                    pk[6] = (__bf16)c.z; pk[7] = (__bf16)c.w;
                    *(v8bf*)&Kbuf[row * KSTRIDE + c0 + j * 8] = pk;
                }
            }
            // ---- stage V chunk transposed -> bf16 [c][kv] ----
            {
                int c0  = 4 * (tid & 31);
                int grp = tid >> 5;
                float4 xv[8];
#pragma unroll
                for (int rr = 0; rr < 8; ++rr) {
                    int l = l0 + grp * 8 + rr;
                    int iy = l / WS, ix = l % WS;
                    int sy = (wy * WS + iy + SHF) % HH;
                    int sx = (wx * WS + ix + SHF) % HH;
                    xv[rr] = *(const float4*)(vbase + ((size_t)sy * HH + sx) * CCH + c0);
                }
#pragma unroll
                for (int cc = 0; cc < 4; ++cc) {
                    v8bf pk;
#pragma unroll
                    for (int rr = 0; rr < 8; ++rr)
                        pk[rr] = (__bf16)(((const float*)&xv[rr])[cc]);
                    *(v8bf*)&VTbuf[(c0 + cc) * VSTRIDE + grp * 8] = pk;
                }
            }
            __syncthreads();

            // ---- S = Q * K^T for two 16-col tiles ----
            v8f s0, s1;
#pragma unroll
            for (int e = 0; e < 8; ++e) { s0[e] = 0.0f; s1[e] = 0.0f; }
#pragma unroll
            for (int ck = 0; ck < 4; ++ck) {
                v16bf bk0 = ld_bfrag16(&Kbuf[ln16 * KSTRIDE + ck * 32 + 16 * half]);
                s0 = __builtin_amdgcn_wmma_f32_16x16x32_bf16(
                        false, qfrag[ck], false, bk0, (short)0, s0, false, false);
                v16bf bk1 = ld_bfrag16(&Kbuf[(16 + ln16) * KSTRIDE + ck * 32 + 16 * half]);
                s1 = __builtin_amdgcn_wmma_f32_16x16x32_bf16(
                        false, qfrag[ck], false, bk1, (short)0, s1, false, false);
            }

            // ---- mask (faithful: mask col = (3l+m) % 576) + online softmax ----
            int j0 = (3 * (l0 + ln16) + m) % LL;
            int j1 = (3 * (l0 + 16 + ln16) + m) % LL;
            int regl0 = region_of(wy, wx, j0);
            int regl1 = region_of(wy, wx, j1);

            float p0[8], p1[8];
#pragma unroll
            for (int r = 0; r < 8; ++r) {
                float a = s0[r] + ((regq[r] == regl0) ? 0.0f : MASKB2);
                float c = s1[r] + ((regq[r] == regl1) ? 0.0f : MASKB2);
                float mx = fmaxf(a, c);
                mx = fmaxf(mx, __shfl_xor(mx, 1, 32));
                mx = fmaxf(mx, __shfl_xor(mx, 2, 32));
                mx = fmaxf(mx, __shfl_xor(mx, 4, 32));
                mx = fmaxf(mx, __shfl_xor(mx, 8, 32));
                float mnew  = fmaxf(mrun[r], mx);
                float alpha = exp2f(mrun[r] - mnew);
                mrun[r] = mnew;
                float e0 = exp2f(a - mnew);
                float e1 = exp2f(c - mnew);
                p0[r] = e0; p1[r] = e1;
                float rs = e0 + e1;
                rs += __shfl_xor(rs, 1, 32);
                rs += __shfl_xor(rs, 2, 32);
                rs += __shfl_xor(rs, 4, 32);
                rs += __shfl_xor(rs, 8, 32);
                lrun[r] = lrun[r] * alpha + rs;
#pragma unroll
                for (int t = 0; t < 8; ++t) acc[t][r] *= alpha;
            }

            // ---- D-layout P -> A-layout via per-wave LDS bounce ----
            float* pw = &Pbuf[wave * 16 * PSTRIDE];
#pragma unroll
            for (int r = 0; r < 8; ++r) {
                pw[(r + 8 * half) * PSTRIDE + ln16]      = p0[r];
                pw[(r + 8 * half) * PSTRIDE + 16 + ln16] = p1[r];
            }
            v16bf pa;
            {
                const float* pr0 = &pw[ln16 * PSTRIDE + 16 * half];
                const float* pr1 = &pw[ln16 * PSTRIDE + 8 + 16 * half];
                float4 x0 = *(const float4*)(pr0);
                float4 x1 = *(const float4*)(pr0 + 4);
                float4 x2 = *(const float4*)(pr1);
                float4 x3 = *(const float4*)(pr1 + 4);
                float t[16] = { x0.x, x0.y, x0.z, x0.w, x1.x, x1.y, x1.z, x1.w,
                                x2.x, x2.y, x2.z, x2.w, x3.x, x3.y, x3.z, x3.w };
#pragma unroll
                for (int e = 0; e < 16; ++e) pa[e] = (__bf16)t[e];
            }

            // ---- O += P * V  (8 column tiles of 16 channels) ----
#pragma unroll
            for (int ct = 0; ct < 8; ++ct) {
                v16bf vb = ld_bfrag16(&VTbuf[(ct * 16 + ln16) * VSTRIDE + 16 * half]);
                acc[ct] = __builtin_amdgcn_wmma_f32_16x16x32_bf16(
                            false, pa, false, vb, (short)0, acc[ct], false, false);
            }
        }
    }

    // ---------------- normalize and scatter-store (roll-back folded in) ----------------
#pragma unroll
    for (int r = 0; r < 8; ++r) {
        int qq = qb + r + 8 * half;
        int iy = qq / WS, ix = qq % WS;
        int sy = (wy * WS + iy + SHF) % HH;
        int sx = (wx * WS + ix + SHF) % HH;
        float* og = out + ((size_t)b * (HH * HH) + (size_t)sy * HH + sx) * CCH;
        float inv = 1.0f / lrun[r];
#pragma unroll
        for (int ct = 0; ct < 8; ++ct)
            og[ct * 16 + ln16] = acc[ct][r] * inv;
    }
}

extern "C" void kernel_launch(void* const* d_in, const int* in_sizes, int n_in,
                              void* d_out, int out_size, void* d_ws, size_t ws_size,
                              hipStream_t stream) {
    (void)in_sizes; (void)n_in; (void)out_size; (void)d_ws; (void)ws_size;
    const float* q = (const float*)d_in[0];
    const float* k = (const float*)d_in[1];
    const float* v = (const float*)d_in[2];
    // d_in[3] = attn_mask (recomputed analytically in-kernel)
    // d_in[4..6] = h, w, num_splits (baked as compile-time constants)
    float* out = (float*)d_out;
    dim3 grid(NWIN * NQT);   // 128 windows * 9 q-tiles = 1152 blocks
    dim3 block(128);         // 4 waves; each owns a 16-row Q tile
    mvum_attn_kernel<<<grid, block, 0, stream>>>(q, k, v, out);
}
